// SchNet_6820408066708
// MI455X (gfx1250) — compile-verified, matrix operand order checked by reading, not run on hardware
//
#include <hip/hip_runtime.h>
#include <hip/hip_bf16.h>
#include <math.h>

// Problem constants (match the reference)
#define N_ATOMS 30000
#define E_EDGES 480000
#define H_DIM   128
#define R_RBF   50
#define R_PAD   64       // pad K for WMMA (multiple of 32), zeros beyond 50
#define NBLK    3
#define NEX_MOL 64
#define CUT_R   10.0f
#define PI_F    3.14159265358979f

typedef __attribute__((ext_vector_type(16))) _Float16 v16h;
typedef __attribute__((ext_vector_type(8)))  float    v8f;

union V16H { v16h v; _Float16 s[16]; };
union V8F  { v8f  v; float    s[8];  };

// ---------------------------------------------------------------------------
// WMMA operand loader.
// 16-bit A-matrix 16x32 (MxK) layout per CDNA5 ISA 7.12.2:
//   lane L: row M = L%16 ; half i holds K = k0 + (i/8)*16 + (L/16)*8 + (i%8)
// B (32x16, KxN) is fed from a TRANSPOSED weight copy Bt[n][k], which gives the
// mirrored per-lane pattern: row = n = col0 + L%16, same K striping.
// ---------------------------------------------------------------------------
template <typename P>
__device__ __forceinline__ v16h load_opd(P base, int ld, int row0, int k0, int lane) {
  int r    = row0 + (lane & 15);
  int koff = k0 + ((lane >> 4) << 3);
  P p = base + (size_t)r * ld + koff;
  V16H u;
#pragma unroll
  for (int g = 0; g < 2; ++g)
#pragma unroll
    for (int j = 0; j < 8; ++j)
      u.s[g * 8 + j] = p[g * 16 + j];
  return u.v;
}

// One wave computes a 16x16 f32 tile: rows row0..row0+15 of A[.,K] times
// cols col0..col0+15 taken from transposed Bt[H][K].
template <typename PA, typename PB>
__device__ __forceinline__ v8f wave_gemm(PA A, int lda, int row0,
                                         PB Bt, int ldb, int col0,
                                         int K, int lane, v8f acc) {
#pragma unroll 4
  for (int k = 0; k < K; k += 32) {
    v16h a = load_opd(A, lda, row0, k, lane);
    v16h b = load_opd(Bt, ldb, col0, k, lane);
    acc = __builtin_amdgcn_wmma_f32_16x16x32_f16(false, a, false, b,
                                                 (short)0, acc, false, false);
  }
  return acc;
}

// D tile element mapping: VGPR r -> M = r + (lane/16)*8, N = lane%16.

// ---------------------------------------------------------------------------
// Small utility kernels
// ---------------------------------------------------------------------------
__global__ void zero_kernel(float* __restrict__ p, int n) {
  int i = blockIdx.x * blockDim.x + threadIdx.x;
  if (i < n) p[i] = 0.0f;
}

__global__ void cvt_f16_kernel(const float* __restrict__ src,
                               _Float16* __restrict__ dst, int n) {
  int i = blockIdx.x * blockDim.x + threadIdx.x;
  if (i < n) dst[i] = (_Float16)src[i];
}

// src: f32 [K,H] row-major  ->  dst: f16 [H,Kpad] row-major (transposed, padded)
__global__ void transpose_convert_kernel(const float* __restrict__ src,
                                         _Float16* __restrict__ dst,
                                         int K, int H, int Kpad) {
  int idx = blockIdx.x * blockDim.x + threadIdx.x;
  if (idx >= H * Kpad) return;
  int hh = idx / Kpad;
  int k  = idx % Kpad;
  float v = (k < K) ? src[(size_t)k * H + hh] : 0.0f;
  dst[idx] = (_Float16)v;
}

// ---------------------------------------------------------------------------
// Edge precompute: distance, cosine cutoff, Gaussian RBF (f16, K padded to 64)
// ---------------------------------------------------------------------------
__global__ void edge_rbf_kernel(const float* __restrict__ pos,
                                const int*   __restrict__ edge_index,
                                _Float16*    __restrict__ rbf,
                                float*       __restrict__ Cbuf) {
  int e = blockIdx.x * blockDim.x + threadIdx.x;
  if (e >= E_EDGES) return;
  int s  = edge_index[e];
  int dd = edge_index[E_EDGES + e];
  float dx = pos[s * 3 + 0] - pos[dd * 3 + 0];
  float dy = pos[s * 3 + 1] - pos[dd * 3 + 1];
  float dz = pos[s * 3 + 2] - pos[dd * 3 + 2];
  float dist = sqrtf(dx * dx + dy * dy + dz * dz + 1e-12f);
  float Cv = (dist < CUT_R) ? 0.5f * (cosf(dist * PI_F / CUT_R) + 1.0f) : 0.0f;
  Cbuf[e] = Cv;
  const float delta = CUT_R / (float)(R_RBF - 1);
  const float coeff = -0.5f / (delta * delta);
  _Float16* row = rbf + (size_t)e * R_PAD;
#pragma unroll
  for (int i = 0; i < R_PAD; ++i) {
    float val = 0.0f;
    if (i < R_RBF) {
      float t = dist - (float)i * delta;
      val = expf(coeff * t * t);
    }
    row[i] = (_Float16)val;
  }
}

// x = embedding[atom_types]; keep f32 master copy + f16 copy for GEMMs
__global__ void embed_kernel(const int*   __restrict__ types,
                             const float* __restrict__ emb,
                             float*       __restrict__ x,
                             _Float16*    __restrict__ xh) {
  int idx = blockIdx.x * blockDim.x + threadIdx.x;
  if (idx >= N_ATOMS * H_DIM) return;
  int n = idx >> 7;
  int c = idx & (H_DIM - 1);
  float v = emb[(size_t)types[n] * H_DIM + c];
  x[idx]  = v;
  xh[idx] = (_Float16)v;
}

// ---------------------------------------------------------------------------
// h = x @ lin1_w   (no bias, f32 out)      grid: N/16 blocks x 256 threads
// ---------------------------------------------------------------------------
__global__ void gemm_lin1_kernel(const _Float16* __restrict__ xh,
                                 const _Float16* __restrict__ wt,
                                 float* __restrict__ h) {
  int lane = threadIdx.x & 31, wave = threadIdx.x >> 5;
  int row0 = blockIdx.x * 16, col0 = wave * 16;
  v8f acc = {};
  acc = wave_gemm(xh, H_DIM, row0, wt, H_DIM, col0, H_DIM, lane, acc);
  V8F u; u.v = acc;
  int n = col0 + (lane & 15);
  int mbase = row0 + ((lane >> 4) << 3);
#pragma unroll
  for (int r = 0; r < 8; ++r)
    h[(size_t)(mbase + r) * H_DIM + n] = u.s[r];
}

// t = tanh(agg @ lin2_w + lin2_b)  -> f16 out
__global__ void gemm_lin2_kernel(const _Float16* __restrict__ ah,
                                 const _Float16* __restrict__ wt,
                                 const float* __restrict__ bias,
                                 _Float16* __restrict__ th) {
  int lane = threadIdx.x & 31, wave = threadIdx.x >> 5;
  int row0 = blockIdx.x * 16, col0 = wave * 16;
  v8f acc = {};
  acc = wave_gemm(ah, H_DIM, row0, wt, H_DIM, col0, H_DIM, lane, acc);
  V8F u; u.v = acc;
  int n = col0 + (lane & 15);
  float bb = bias[n];
  int mbase = row0 + ((lane >> 4) << 3);
#pragma unroll
  for (int r = 0; r < 8; ++r)
    th[(size_t)(mbase + r) * H_DIM + n] = (_Float16)tanhf(u.s[r] + bb);
}

// x += t @ blk_w + blk_b ; refresh f16 copy of x
__global__ void gemm_blk_kernel(const _Float16* __restrict__ th,
                                const _Float16* __restrict__ wt,
                                const float* __restrict__ bias,
                                float* __restrict__ x,
                                _Float16* __restrict__ xh) {
  int lane = threadIdx.x & 31, wave = threadIdx.x >> 5;
  int row0 = blockIdx.x * 16, col0 = wave * 16;
  v8f acc = {};
  acc = wave_gemm(th, H_DIM, row0, wt, H_DIM, col0, H_DIM, lane, acc);
  V8F u; u.v = acc;
  int n = col0 + (lane & 15);
  float bb = bias[n];
  int mbase = row0 + ((lane >> 4) << 3);
#pragma unroll
  for (int r = 0; r < 8; ++r) {
    size_t idx = (size_t)(mbase + r) * H_DIM + n;
    float v = x[idx] + u.s[r] + bb;
    x[idx]  = v;
    xh[idx] = (_Float16)v;
  }
}

// ---------------------------------------------------------------------------
// Fused cfconv edge kernel, one 16-edge tile per 256-thread block:
//   W = tanh(rbf @ fw1 + fb1) @ fw2 + fb2 ; W *= C[e]
//   agg[dst[e]] += h[src[e]] * W[e]     (atomic scatter)
// GEMM1 column tiles are exchanged through LDS (f16 16x128 tile) so each wave
// can run GEMM2 over the full K=128.
// ---------------------------------------------------------------------------
__global__ void cfconv_kernel(const _Float16* __restrict__ rbf,
                              const _Float16* __restrict__ w1t,
                              const float*    __restrict__ b1,
                              const _Float16* __restrict__ w2t,
                              const float*    __restrict__ b2,
                              const float*    __restrict__ Cbuf,
                              const int*      __restrict__ src,
                              const int*      __restrict__ dst,
                              const float*    __restrict__ h,
                              float*          __restrict__ agg) {
  __shared__ _Float16 Ash[16 * 136];          // 16 edges x 128 ch, padded ld
  int lane = threadIdx.x & 31, wave = threadIdx.x >> 5;
  int e0   = blockIdx.x * 16;
  int col0 = wave * 16;

  // GEMM1: rbf[e0..e0+15, 0..63] @ filt_w1 -> this wave's 16 channels
  v8f acc1 = {};
  acc1 = wave_gemm(rbf, R_PAD, e0, w1t, R_PAD, col0, R_PAD, lane, acc1);
  V8F u1; u1.v = acc1;
  int n = col0 + (lane & 15);
  float bb1 = b1[n];
  int mloc = (lane >> 4) << 3;
#pragma unroll
  for (int r = 0; r < 8; ++r)
    Ash[(mloc + r) * 136 + n] = (_Float16)tanhf(u1.s[r] + bb1);
  __syncthreads();

  // GEMM2: tanh-tile (LDS) @ filt_w2 over K=128
  v8f acc2 = {};
  acc2 = wave_gemm((const _Float16*)Ash, 136, 0, w2t, H_DIM, col0, H_DIM, lane, acc2);
  V8F u2; u2.v = acc2;
  float bb2 = b2[n];
#pragma unroll
  for (int r = 0; r < 8; ++r) {
    int e = e0 + mloc + r;
    float wv = (u2.s[r] + bb2) * Cbuf[e];
    int s  = src[e];
    int dd = dst[e];
    float msg = wv * h[(size_t)s * H_DIM + n];
    atomicAdd(&agg[(size_t)dd * H_DIM + n], msg);
  }
}

// ---------------------------------------------------------------------------
// Output head: e = tanh(x @ out_w1 + b1) @ out_w2 + b2 ; energy[batch] += e
// One wave per atom; each lane owns 2 of the 64 hidden channels.
// ---------------------------------------------------------------------------
__global__ void output_kernel(const float* __restrict__ x,
                              const float* __restrict__ w1,
                              const float* __restrict__ b1,
                              const float* __restrict__ w2,
                              const float* __restrict__ b2,
                              const int*   __restrict__ batch,
                              float*       __restrict__ energy) {
  int lane = threadIdx.x & 31;
  int atom = blockIdx.x * (blockDim.x >> 5) + (threadIdx.x >> 5);
  if (atom >= N_ATOMS) return;
  const float* xr = x + (size_t)atom * H_DIM;
  float partial = 0.0f;
#pragma unroll
  for (int cc = 0; cc < 2; ++cc) {
    int c = lane + cc * 32;
    float acc = b1[c];
    for (int k = 0; k < H_DIM; ++k)
      acc += xr[k] * w1[(size_t)k * 64 + c];
    partial += tanhf(acc) * w2[c];
  }
#pragma unroll
  for (int off = 16; off > 0; off >>= 1)
    partial += __shfl_xor(partial, off, 32);
  if (lane == 0)
    atomicAdd(&energy[batch[atom]], partial + b2[0]);
}

// ---------------------------------------------------------------------------
// Host orchestration
// ---------------------------------------------------------------------------
extern "C" void kernel_launch(void* const* d_in, const int* in_sizes, int n_in,
                              void* d_out, int out_size, void* d_ws, size_t ws_size,
                              hipStream_t stream) {
  (void)in_sizes; (void)n_in; (void)out_size; (void)ws_size;
  const float* pos        = (const float*)d_in[0];
  const int*   atom_types = (const int*)  d_in[1];
  const int*   edge_index = (const int*)  d_in[2];
  const int*   batch      = (const int*)  d_in[3];
  const float* embedding  = (const float*)d_in[4];
  const float* filt_w1    = (const float*)d_in[5];
  const float* filt_b1    = (const float*)d_in[6];
  const float* filt_w2    = (const float*)d_in[7];
  const float* filt_b2    = (const float*)d_in[8];
  const float* lin1_w     = (const float*)d_in[9];
  const float* lin2_w     = (const float*)d_in[10];
  const float* lin2_b     = (const float*)d_in[11];
  const float* blk_w      = (const float*)d_in[12];
  const float* blk_b      = (const float*)d_in[13];
  const float* out_w1     = (const float*)d_in[14];
  const float* out_b1     = (const float*)d_in[15];
  const float* out_w2     = (const float*)d_in[16];
  const float* out_b2     = (const float*)d_in[17];
  float* energy = (float*)d_out;

  // Workspace partitioning
  char* ws = (char*)d_ws;
  size_t off = 0;
  auto walloc = [&](size_t bytes) -> void* {
    void* p = ws + off;
    off = (off + bytes + 255) & ~(size_t)255;
    return p;
  };
  _Float16* rbf  = (_Float16*)walloc((size_t)E_EDGES * R_PAD * sizeof(_Float16));
  float*    Cbuf = (float*)   walloc((size_t)E_EDGES * sizeof(float));
  float*    x    = (float*)   walloc((size_t)N_ATOMS * H_DIM * sizeof(float));
  _Float16* xh   = (_Float16*)walloc((size_t)N_ATOMS * H_DIM * sizeof(_Float16));
  float*    hbuf = (float*)   walloc((size_t)N_ATOMS * H_DIM * sizeof(float));
  float*    agg  = (float*)   walloc((size_t)N_ATOMS * H_DIM * sizeof(float));
  _Float16* aggh = (_Float16*)walloc((size_t)N_ATOMS * H_DIM * sizeof(_Float16));
  _Float16* th   = (_Float16*)walloc((size_t)N_ATOMS * H_DIM * sizeof(_Float16));
  _Float16* w1t  = (_Float16*)walloc((size_t)NBLK * H_DIM * R_PAD * sizeof(_Float16));
  _Float16* w2t  = (_Float16*)walloc((size_t)NBLK * H_DIM * H_DIM * sizeof(_Float16));
  _Float16* l1t  = (_Float16*)walloc((size_t)NBLK * H_DIM * H_DIM * sizeof(_Float16));
  _Float16* l2t  = (_Float16*)walloc((size_t)NBLK * H_DIM * H_DIM * sizeof(_Float16));
  _Float16* bkt  = (_Float16*)walloc((size_t)NBLK * H_DIM * H_DIM * sizeof(_Float16));

  const int TPB = 256;
  const int NH  = N_ATOMS * H_DIM;   // 3,840,000 (divisible by 256)

  // Transposed f16 weight copies
  for (int b = 0; b < NBLK; ++b) {
    transpose_convert_kernel<<<(H_DIM * R_PAD + TPB - 1) / TPB, TPB, 0, stream>>>(
        filt_w1 + (size_t)b * R_RBF * H_DIM, w1t + (size_t)b * H_DIM * R_PAD,
        R_RBF, H_DIM, R_PAD);
    transpose_convert_kernel<<<(H_DIM * H_DIM + TPB - 1) / TPB, TPB, 0, stream>>>(
        filt_w2 + (size_t)b * H_DIM * H_DIM, w2t + (size_t)b * H_DIM * H_DIM,
        H_DIM, H_DIM, H_DIM);
    transpose_convert_kernel<<<(H_DIM * H_DIM + TPB - 1) / TPB, TPB, 0, stream>>>(
        lin1_w + (size_t)b * H_DIM * H_DIM, l1t + (size_t)b * H_DIM * H_DIM,
        H_DIM, H_DIM, H_DIM);
    transpose_convert_kernel<<<(H_DIM * H_DIM + TPB - 1) / TPB, TPB, 0, stream>>>(
        lin2_w + (size_t)b * H_DIM * H_DIM, l2t + (size_t)b * H_DIM * H_DIM,
        H_DIM, H_DIM, H_DIM);
    transpose_convert_kernel<<<(H_DIM * H_DIM + TPB - 1) / TPB, TPB, 0, stream>>>(
        blk_w + (size_t)b * H_DIM * H_DIM, bkt + (size_t)b * H_DIM * H_DIM,
        H_DIM, H_DIM, H_DIM);
  }

  // Edge geometry + RBF, atom embedding
  edge_rbf_kernel<<<E_EDGES / TPB, TPB, 0, stream>>>(pos, edge_index, rbf, Cbuf);
  embed_kernel<<<NH / TPB, TPB, 0, stream>>>(atom_types, embedding, x, xh);

  // Interaction blocks
  for (int b = 0; b < NBLK; ++b) {
    zero_kernel<<<NH / TPB, TPB, 0, stream>>>(agg, NH);
    gemm_lin1_kernel<<<N_ATOMS / 16, TPB, 0, stream>>>(
        xh, l1t + (size_t)b * H_DIM * H_DIM, hbuf);
    cfconv_kernel<<<E_EDGES / 16, TPB, 0, stream>>>(
        rbf, w1t + (size_t)b * H_DIM * R_PAD, filt_b1 + (size_t)b * H_DIM,
        w2t + (size_t)b * H_DIM * H_DIM, filt_b2 + (size_t)b * H_DIM,
        Cbuf, edge_index /*src*/, edge_index + E_EDGES /*dst*/, hbuf, agg);
    cvt_f16_kernel<<<NH / TPB, TPB, 0, stream>>>(agg, aggh, NH);
    gemm_lin2_kernel<<<N_ATOMS / 16, TPB, 0, stream>>>(
        aggh, l2t + (size_t)b * H_DIM * H_DIM, lin2_b + (size_t)b * H_DIM, th);
    gemm_blk_kernel<<<N_ATOMS / 16, TPB, 0, stream>>>(
        th, bkt + (size_t)b * H_DIM * H_DIM, blk_b + (size_t)b * H_DIM, x, xh);
  }

  // Output head + per-molecule reduction
  zero_kernel<<<1, NEX_MOL, 0, stream>>>(energy, NEX_MOL);
  output_kernel<<<N_ATOMS / 8, TPB, 0, stream>>>(
      x, out_w1, out_b1, out_w2, out_b2, batch, energy);
}
